// TransformerNet_ray_1769526526171
// MI455X (gfx1250) — matrix operand order repaired and checked by reading, not compiled, not run on hardware
//
#include <hip/hip_runtime.h>
#include <cstdint>
#include <cstddef>

// ---------------------------------------------------------------------------
// Types for CDNA5 WMMA (wave32): v_wmma_f32_16x16x32_bf16
// ---------------------------------------------------------------------------
typedef __attribute__((ext_vector_type(16))) __bf16 v16bf;
typedef __attribute__((ext_vector_type(8)))  float  v8f;

#define HD    128   // H*D hidden
#define COLS  512   // q|k|v|skip concatenated
#define RSQRT_D 0.1767766953f  // 1/sqrt(32)

union FragU { v16bf v; uint4 q[2]; };

__device__ __forceinline__ uint16_t f2bf(float f) {
    uint32_t u = __float_as_uint(f);
    uint32_t r = (u + 0x7FFFu + ((u >> 16) & 1u)) >> 16;   // RNE
    return (uint16_t)r;
}

__device__ __forceinline__ void atomicMaxF(float* addr, float v) {
    // order-preserving trick; mbuf initialized to -inf (0xFF800000)
    if (v >= 0.0f) atomicMax((int*)addr, __float_as_int(v));
    else           atomicMin((unsigned int*)addr, __float_as_uint(v));
}

__device__ __forceinline__ void atomicAddF(float* addr, float v) {
    __hip_atomic_fetch_add(addr, v, __ATOMIC_RELAXED, __HIP_MEMORY_SCOPE_AGENT);
}

// ---------------------------------------------------------------------------
// Convert f32 activations -> bf16 (layer-0 input)
// ---------------------------------------------------------------------------
__global__ void cvt_bf16_kernel(const float* __restrict__ x,
                                uint16_t* __restrict__ xb, int total) {
    int i = blockIdx.x * blockDim.x + threadIdx.x;
    if (i < total) xb[i] = f2bf(x[i]);
}

// ---------------------------------------------------------------------------
// Repack [din x 128] x4 weight matrices into the per-lane-contiguous B-fragment
// layout for v_wmma_f32_16x16x32_bf16.
//   B 32x16 bf16 fragment: lane L -> column (L&15); halves j=0..15 -> K = 16*(L>>4)+j
//   wpack[((ct*nkt + kt)*32 + lane)*16 + j]
// ---------------------------------------------------------------------------
__global__ void pack_w_kernel(const float* __restrict__ Wq, const float* __restrict__ Wk,
                              const float* __restrict__ Wv, const float* __restrict__ Ws,
                              uint16_t* __restrict__ wpack, int din) {
    int idx = blockIdx.x * blockDim.x + threadIdx.x;
    int total = COLS * din;
    if (idx >= total) return;
    int nkt  = din >> 5;
    int j    = idx & 15;
    int lane = (idx >> 4) & 31;
    int t    = idx >> 9;          // = ct*nkt + kt
    int kt   = t % nkt;
    int ct   = t / nkt;
    int K    = kt * 32 + 16 * (lane >> 4) + j;
    int col  = ct * 16 + (lane & 15);
    int mat  = col >> 7;
    int c    = col & 127;
    const float* W = (mat == 0) ? Wq : (mat == 1) ? Wk : (mat == 2) ? Wv : Ws;
    wpack[idx] = f2bf(W[(size_t)K * HD + c]);
}

__global__ void pack_b_kernel(const float* __restrict__ bq, const float* __restrict__ bk,
                              const float* __restrict__ bv, const float* __restrict__ bs,
                              float* __restrict__ bpack) {
    int idx = blockIdx.x * blockDim.x + threadIdx.x;
    if (idx >= COLS) return;
    int mat = idx >> 7, c = idx & 127;
    const float* b = (mat == 0) ? bq : (mat == 1) ? bk : (mat == 2) ? bv : bs;
    bpack[idx] = b[c];
}

// ---------------------------------------------------------------------------
// Init per-layer segment buffers: agg=0, m=-inf, den=0
// ---------------------------------------------------------------------------
__global__ void init_kernel(float* __restrict__ mbuf, float* __restrict__ den,
                            float* __restrict__ agg, int n) {
    int i = blockIdx.x * blockDim.x + threadIdx.x;
    if (i < n * HD) agg[i] = 0.0f;
    if (i < n * 4) { mbuf[i] = __uint_as_float(0xFF800000u); den[i] = 0.0f; }
}

// ---------------------------------------------------------------------------
// Fused QKVS projection: qkvs[N x 512] = Xb[N x din] @ Wpack + bias
// One wave per 16x16 output tile; 8 waves/block -> 128 columns per block.
// grid = (4, N/16), block = 256 (wave32 -> 8 waves)
// ---------------------------------------------------------------------------
__global__ void gemm_qkvs_kernel(const uint16_t* __restrict__ xb,
                                 const uint16_t* __restrict__ wpack,
                                 const float* __restrict__ bpack,
                                 float* __restrict__ qkvs,
                                 int n, int din) {
    const int lane = threadIdx.x & 31;
    const int ct   = blockIdx.x * 8 + (threadIdx.x >> 5);   // column tile 0..31
    const int tr   = blockIdx.y;                            // row tile
    const int row  = tr * 16 + (lane & 15);
    const int kb   = (lane >> 4) * 16;                      // A-fragment K sub-base
    const int nkt  = din >> 5;

    v8f acc = {0.f, 0.f, 0.f, 0.f, 0.f, 0.f, 0.f, 0.f};
    const uint16_t* arow = xb + (size_t)row * din;

    for (int kt = 0; kt < nkt; ++kt) {
        FragU a, b;
        const uint16_t* ap = arow + kt * 32;
        a.q[0] = *(const uint4*)(ap + kb);        // K = kb .. kb+7
        a.q[1] = *(const uint4*)(ap + kb + 8);    // K = kb+8 .. kb+15
        const uint16_t* bp = wpack + ((size_t)(ct * nkt + kt) * 32 + lane) * 16;
        b.q[0] = *(const uint4*)(bp);
        b.q[1] = *(const uint4*)(bp + 8);
        acc = __builtin_amdgcn_wmma_f32_16x16x32_bf16(
            false, a.v, false, b.v, (short)0, acc, false, false);
    }

    const int col   = ct * 16 + (lane & 15);
    const float bia = bpack[col];
    const int rbase = tr * 16 + 8 * (lane >> 4);
    float* outp = qkvs + (size_t)rbase * COLS + col;
#pragma unroll
    for (int r = 0; r < 8; ++r)
        outp[(size_t)r * COLS] = acc[r] + bia;
}

// ---------------------------------------------------------------------------
// Per-edge attention score + segment max (thread per edge, 4 heads each)
// ---------------------------------------------------------------------------
__global__ void score_kernel(const float* __restrict__ qkvs,
                             const int* __restrict__ src, const int* __restrict__ dst,
                             float* __restrict__ sbuf, float* __restrict__ mbuf, int E) {
    int e = blockIdx.x * blockDim.x + threadIdx.x;
    if (e >= E) return;
    int s = src[e], d = dst[e];
    const float4* qp = (const float4*)(qkvs + (size_t)d * COLS);         // q cols 0..127
    const float4* kp = (const float4*)(qkvs + (size_t)s * COLS + HD);    // k cols 128..255
#pragma unroll
    for (int h = 0; h < 4; ++h) {
        float acc = 0.0f;
#pragma unroll
        for (int i = 0; i < 8; ++i) {
            float4 a = qp[h * 8 + i], b = kp[h * 8 + i];
            acc += a.x * b.x + a.y * b.y + a.z * b.z + a.w * b.w;
        }
        acc *= RSQRT_D;
        sbuf[(size_t)e * 4 + h] = acc;
        atomicMaxF(mbuf + (size_t)d * 4 + h, acc);
    }
}

// ---------------------------------------------------------------------------
// exp(s - m[dst]) in-place + segment sum of denominators (thread per edge-head)
// ---------------------------------------------------------------------------
__global__ void exp_kernel(const int* __restrict__ dst, const float* __restrict__ mbuf,
                           float* __restrict__ sbuf, float* __restrict__ den, int E) {
    int idx = blockIdx.x * blockDim.x + threadIdx.x;
    if (idx >= E * 4) return;
    int e = idx >> 2, h = idx & 3;
    int d = dst[e];
    float ev = __expf(sbuf[idx] - mbuf[(size_t)d * 4 + h]);
    sbuf[idx] = ev;
    atomicAddF(den + (size_t)d * 4 + h, ev);
}

// ---------------------------------------------------------------------------
// agg[dst] += alpha * v[src]  (thread per edge-head, 32 channels)
// ---------------------------------------------------------------------------
__global__ void agg_kernel(const float* __restrict__ qkvs,
                           const int* __restrict__ src, const int* __restrict__ dst,
                           const float* __restrict__ sbuf, const float* __restrict__ den,
                           float* __restrict__ agg, int E) {
    int idx = blockIdx.x * blockDim.x + threadIdx.x;
    if (idx >= E * 4) return;
    int e = idx >> 2, h = idx & 3;
    int s = src[e], d = dst[e];
    float alpha = sbuf[idx] / den[(size_t)d * 4 + h];
    const float* vp = qkvs + (size_t)s * COLS + 2 * HD + h * 32;  // v cols 256..383
    float* op = agg + (size_t)d * HD + h * 32;
#pragma unroll
    for (int i = 0; i < 32; ++i)
        atomicAddF(op + i, alpha * vp[i]);
}

// ---------------------------------------------------------------------------
// h = relu(agg + skip); also emit bf16 copy for next layer's WMMA input
// ---------------------------------------------------------------------------
__global__ void finalize_kernel(const float* __restrict__ qkvs,
                                const float* __restrict__ agg,
                                float* __restrict__ hbuf, uint16_t* __restrict__ xb,
                                int total) {
    int idx = blockIdx.x * blockDim.x + threadIdx.x;
    if (idx >= total) return;
    int n = idx >> 7, c = idx & 127;
    float v = agg[idx] + qkvs[(size_t)n * COLS + 3 * HD + c];   // skip cols 384..511
    v = fmaxf(v, 0.0f);
    hbuf[idx] = v;
    xb[idx]   = f2bf(v);
}

// ---------------------------------------------------------------------------
// out = sigmoid(h @ Wf + bf)   (thread per node)
// ---------------------------------------------------------------------------
__global__ void head_kernel(const float* __restrict__ hbuf, const float* __restrict__ Wf,
                            const float* __restrict__ bf, float* __restrict__ out, int n) {
    int i = blockIdx.x * blockDim.x + threadIdx.x;
    if (i >= n) return;
    float acc = bf[0];
    const float* hp = hbuf + (size_t)i * HD;
#pragma unroll 8
    for (int c = 0; c < HD; ++c) acc += hp[c] * Wf[c];
    out[i] = 1.0f / (1.0f + __expf(-acc));
}

// ---------------------------------------------------------------------------
// Host-side launcher
// ---------------------------------------------------------------------------
extern "C" void kernel_launch(void* const* d_in, const int* in_sizes, int n_in,
                              void* d_out, int out_size, void* d_ws, size_t ws_size,
                              hipStream_t stream) {
    const float* x  = (const float*)d_in[0];
    const int*   ei = (const int*)d_in[1];
    const int N = in_sizes[0] / 32;
    const int E = in_sizes[1] / 2;
    const int* src = ei;
    const int* dst = ei + E;
    const float* Wf = (const float*)d_in[26];
    const float* bf = (const float*)d_in[27];

    // Workspace carve-up (256B aligned)
    char* ws = (char*)d_ws;
    size_t off = 0;
    auto carve = [&](size_t bytes) -> void* {
        void* p = ws + off;
        off = (off + bytes + 255) & ~(size_t)255;
        return p;
    };
    uint16_t* xb    = (uint16_t*)carve((size_t)N * HD * sizeof(uint16_t));
    float*    qkvs  = (float*)   carve((size_t)N * COLS * sizeof(float));
    float*    sbuf  = (float*)   carve((size_t)E * 4 * sizeof(float));
    float*    mbuf  = (float*)   carve((size_t)N * 4 * sizeof(float));
    float*    den   = (float*)   carve((size_t)N * 4 * sizeof(float));
    float*    agg   = (float*)   carve((size_t)N * HD * sizeof(float));
    uint16_t* wpack = (uint16_t*)carve((size_t)HD * COLS * sizeof(uint16_t));
    float*    bpack = (float*)   carve((size_t)COLS * sizeof(float));
    float*    hbuf  = (float*)   carve((size_t)N * HD * sizeof(float));

    const int TB = 256;
    auto blocks = [&](long long t) { return (int)((t + TB - 1) / TB); };

    // Layer-0 input -> bf16
    cvt_bf16_kernel<<<blocks((long long)N * 32), TB, 0, stream>>>(x, xb, N * 32);

    for (int l = 0; l < 3; ++l) {
        const int din = (l == 0) ? 32 : 128;
        const float* Wq = (const float*)d_in[2 + l * 8 + 0];
        const float* bq = (const float*)d_in[2 + l * 8 + 1];
        const float* Wk = (const float*)d_in[2 + l * 8 + 2];
        const float* bk = (const float*)d_in[2 + l * 8 + 3];
        const float* Wv = (const float*)d_in[2 + l * 8 + 4];
        const float* bv = (const float*)d_in[2 + l * 8 + 5];
        const float* Ws = (const float*)d_in[2 + l * 8 + 6];
        const float* bs = (const float*)d_in[2 + l * 8 + 7];

        pack_w_kernel<<<blocks((long long)COLS * din), TB, 0, stream>>>(Wq, Wk, Wv, Ws, wpack, din);
        pack_b_kernel<<<blocks(COLS), TB, 0, stream>>>(bq, bk, bv, bs, bpack);
        init_kernel<<<blocks((long long)N * HD), TB, 0, stream>>>(mbuf, den, agg, N);

        dim3 ggrid(4, N / 16);   // N = 50000 = 3125 * 16
        gemm_qkvs_kernel<<<ggrid, TB, 0, stream>>>(xb, wpack, bpack, qkvs, N, din);

        score_kernel<<<blocks(E), TB, 0, stream>>>(qkvs, src, dst, sbuf, mbuf, E);
        exp_kernel<<<blocks((long long)E * 4), TB, 0, stream>>>(dst, mbuf, sbuf, den, E);
        agg_kernel<<<blocks((long long)E * 4), TB, 0, stream>>>(qkvs, src, dst, sbuf, den, agg, E);
        finalize_kernel<<<blocks((long long)N * HD), TB, 0, stream>>>(qkvs, agg, hbuf, xb, N * HD);
    }

    head_kernel<<<blocks(N), TB, 0, stream>>>(hbuf, Wf, bf, (float*)d_out, N);
}
